// MachineAwareAdapter_2516850835769
// MI455X (gfx1250) — matrix operand order repaired and checked by reading, not compiled, not run on hardware
//
#include <hip/hip_runtime.h>

// ---------------------------------------------------------------------------
// MachineAwareAdapter: out[i] = x[i] + relu(x[i]@W1[e] + b1[e])@W2[e] + b2[e]
// e = expert_ids[i].  B=32768, D=768, H=64, E=7 (derived at launch).
// Plan: bin rows by expert, then 64-row blocks do bf16 WMMA GEMMs from LDS.
// ---------------------------------------------------------------------------

#define DM   768              // d_model
#define HB   64               // bottleneck
#define ROWS 64               // rows per block
#define NTHR 256              // 8 waves (wave32)

#define XP   (DM + 8)         // x LDS pitch (bf16 elems), 776
#define W1P  (DM + 8)         // W1^T pitch: rows are n(0..63), cols k(0..767)
#define W2P  (HB + 8)         // W2^T pitch: rows are n(0..767), cols k(0..63)
#define HP   (HB + 8)         // h pitch

// LDS partition (in ushorts), then 64 ints of row indices
#define X_ELEMS   (ROWS * XP)
#define W1_ELEMS  (HB * W1P)
#define W2_ELEMS  (DM * W2P)
#define H_ELEMS   (ROWS * HP)
#define SMEM_BYTES ((X_ELEMS + W1_ELEMS + W2_ELEMS + H_ELEMS) * 2 + ROWS * 4)

typedef __attribute__((ext_vector_type(16))) __bf16        v16bf;
typedef __attribute__((ext_vector_type(8)))  float         v8f;
typedef __attribute__((ext_vector_type(4)))  unsigned int  v4u;

union Frag { v4u q[2]; v16bf v; };

__device__ __forceinline__ unsigned short f2bf(float f) {
  unsigned int u = __float_as_uint(f);
  u += 0x7FFFu + ((u >> 16) & 1u);          // round-to-nearest-even
  return (unsigned short)(u >> 16);
}

// A-matrix 16x32 bf16 fragment (ISA 7.12.2): halfs 0-7 = K kb..kb+7,
// halfs 8-15 = K kb+16..kb+23, kb = 8*(lane>=16). p points at (row, k0+kb).
__device__ __forceinline__ v16bf ldfragA(const unsigned short* p) {
  Frag f;
  f.q[0] = *reinterpret_cast<const v4u*>(p);
  f.q[1] = *reinterpret_cast<const v4u*>(p + 16);
  return f.v;
}

// B-matrix 32x16 bf16 fragment: per lane, col = lane&15, halfs 0-7 = K kb..kb+7,
// halfs 8-15 = K kb+16..kb+23, kb = 8*(lane>=16) (same K order as A).
__device__ __forceinline__ v16bf ldfragB(const unsigned short* p) {
  Frag f;
  f.q[0] = *reinterpret_cast<const v4u*>(p);
  f.q[1] = *reinterpret_cast<const v4u*>(p + 16);
  return f.v;
}

// ----------------------------- binning kernels -----------------------------
// ws layout (ints): [0..7]=counts  [8..15]=offsets  [16..23]=cursors
//                   [24..24+B)=row indices grouped by expert
__global__ void k_init(int* ws) {
  if (threadIdx.x < 24) ws[threadIdx.x] = 0;
}

__global__ void k_count(const int* __restrict__ ids, int* ws, int B, int E) {
  int i = blockIdx.x * blockDim.x + threadIdx.x;
  if (i < B) {
    int e = ids[i];
    if ((unsigned)e < (unsigned)E) atomicAdd(&ws[e], 1);
  }
}

__global__ void k_scan(int* ws, int E) {
  if (threadIdx.x == 0) {
    int s = 0;
    for (int e = 0; e < E; ++e) { ws[8 + e] = s; s += ws[e]; }
  }
}

__global__ void k_scatter(const int* __restrict__ ids, int* ws, int B, int E) {
  int i = blockIdx.x * blockDim.x + threadIdx.x;
  if (i < B) {
    int e = ids[i];
    if ((unsigned)e < (unsigned)E) {
      int p = atomicAdd(&ws[16 + e], 1);
      ws[24 + ws[8 + e] + p] = i;
    }
  }
}

// ------------------------------- main kernel -------------------------------
__global__ __launch_bounds__(NTHR)
void adapter_kernel(const float* __restrict__ x,
                    const float* __restrict__ W1, const float* __restrict__ bias1,
                    const float* __restrict__ W2, const float* __restrict__ bias2,
                    float* __restrict__ out,
                    const int* __restrict__ ws, int B, int E) {
  extern __shared__ unsigned char smem_raw[];
  unsigned short* xs  = (unsigned short*)smem_raw;      // [ROWS][XP]  bf16 x
  unsigned short* w1t = xs  + X_ELEMS;                  // [HB][W1P]   W1^T
  unsigned short* w2t = w1t + W1_ELEMS;                 // [DM][W2P]   W2^T
  unsigned short* hbf = w2t + W2_ELEMS;                 // [ROWS][HP]  h
  int* ridx = (int*)(hbf + H_ELEMS);                    // [ROWS] row ids

  const int tid = threadIdx.x;

  // Map flat block id -> (expert e, 64-row tile blk) from device-side counts.
  int e = -1, blk = 0, accb = 0;
  for (int t = 0; t < E; ++t) {
    int nb = (ws[t] + ROWS - 1) >> 6;
    if (e < 0 && (int)blockIdx.x < accb + nb) { e = t; blk = blockIdx.x - accb; }
    accb += nb;
  }
  if (e < 0) return;
  const int cnt  = ws[e];
  const int base = ws[8 + e] + blk * ROWS;

  if (tid < ROWS) {
    int local = blk * ROWS + tid;
    ridx[tid] = (local < cnt) ? ws[24 + base - blk * ROWS + local] : -1;
  }
  __syncthreads();

  // ---- stage x rows (gather, fp32 -> bf16) ----
  const float* w1g = W1 + (size_t)e * DM * HB;
  const float* w2g = W2 + (size_t)e * HB * DM;
  for (int rr = 0; rr < ROWS; ++rr) {
    int grow = ridx[rr];
    const float* xr = x + (size_t)(grow < 0 ? 0 : grow) * DM;
    for (int c = tid; c < DM; c += NTHR)
      xs[rr * XP + c] = (grow >= 0) ? f2bf(xr[c]) : (unsigned short)0;
  }
  // ---- stage W1^T: w1t[n][k] = W1[e][k][n] ----
  for (int i = tid; i < DM * HB; i += NTHR) {
    int k = i >> 6, n = i & (HB - 1);
    w1t[n * W1P + k] = f2bf(w1g[i]);
  }
  // ---- stage W2^T: w2t[n][k] = W2[e][k][n] ----
  for (int i = tid; i < HB * DM; i += NTHR) {
    int k = i / DM, n = i - k * DM;
    w2t[n * W2P + k] = f2bf(w2g[i]);
  }
  __syncthreads();

  const int w    = tid >> 5;
  const int lane = tid & 31;
  const int lrow = lane & 15;
  const int lhi  = lane >> 4;
  const int kbA  = lhi * 8;     // A-fragment K base
  const int kbB  = lhi * 8;     // B-fragment K base
  const int m0   = (w >> 1) * 16;
  const int n0   = (w & 1) * 32;

  // ================= GEMM1: h[64x64] = relu(x @ W1 + b1) =================
  v8f acc0 = {}, acc1 = {};
  for (int kk = 0; kk < DM; kk += 32) {
    v16bf a   = ldfragA(xs  + (m0 + lrow) * XP + kk + kbA);
    v16bf bf0 = ldfragB(w1t + (n0 + lrow) * W1P + kk + kbB);
    v16bf bf1 = ldfragB(w1t + (n0 + 16 + lrow) * W1P + kk + kbB);
    acc0 = __builtin_amdgcn_wmma_f32_16x16x32_bf16(false, a, false, bf0,
                                                   (short)0, acc0, false, false);
    acc1 = __builtin_amdgcn_wmma_f32_16x16x32_bf16(false, a, false, bf1,
                                                   (short)0, acc1, false, false);
  }
  {
    const int na = n0 + lrow, nb = n0 + 16 + lrow;
    const float ba = bias1[e * HB + na];
    const float bb = bias1[e * HB + nb];
#pragma unroll
    for (int i = 0; i < 8; ++i) {
      int m = m0 + lhi * 8 + i;
      float v0 = acc0[i] + ba; v0 = v0 > 0.f ? v0 : 0.f;
      float v1 = acc1[i] + bb; v1 = v1 > 0.f ? v1 : 0.f;
      hbf[m * HP + na] = f2bf(v0);
      hbf[m * HP + nb] = f2bf(v1);
    }
  }
  __syncthreads();

  // ============ GEMM2: y[64x768] = h @ W2 + b2 ; out = x + y ============
  v16bf a20 = ldfragA(hbf + (m0 + lrow) * HP + 0  + kbA);
  v16bf a21 = ldfragA(hbf + (m0 + lrow) * HP + 32 + kbA);
  for (int j = 0; j < (DM / 16) / 2; ++j) {
    const int n0y = ((w & 1) + 2 * j) * 16;
    const unsigned short* pb = w2t + (n0y + lrow) * W2P + kbB;
    v16bf bf0 = ldfragB(pb);        // K = 0..31
    v16bf bf1 = ldfragB(pb + 32);   // K = 32..63
    v8f acc = {};
    acc = __builtin_amdgcn_wmma_f32_16x16x32_bf16(false, a20, false, bf0,
                                                  (short)0, acc, false, false);
    acc = __builtin_amdgcn_wmma_f32_16x16x32_bf16(false, a21, false, bf1,
                                                  (short)0, acc, false, false);
    const int n = n0y + lrow;
    const float bias = bias2[e * DM + n];
#pragma unroll
    for (int i = 0; i < 8; ++i) {
      int m = m0 + lhi * 8 + i;
      int grow = ridx[m];
      if (grow >= 0) {
        size_t off = (size_t)grow * DM + n;
        out[off] = x[off] + acc[i] + bias;   // residual x hits L2 (96MB < 192MB)
      }
    }
  }
}

// ------------------------------- launcher ----------------------------------
extern "C" void kernel_launch(void* const* d_in, const int* in_sizes, int n_in,
                              void* d_out, int out_size, void* d_ws, size_t ws_size,
                              hipStream_t stream) {
  const float* x   = (const float*)d_in[0];
  const int*   ids = (const int*)  d_in[1];
  const float* W1  = (const float*)d_in[2];
  const float* b1  = (const float*)d_in[3];
  const float* W2  = (const float*)d_in[4];
  const float* b2  = (const float*)d_in[5];
  float* out = (float*)d_out;

  const int B = in_sizes[0] / DM;
  const int E = in_sizes[2] / (DM * HB);
  int* ws = (int*)d_ws;

  k_init<<<1, 32, 0, stream>>>(ws);
  k_count<<<(B + 255) / 256, 256, 0, stream>>>(ids, ws, B, E);
  k_scan<<<1, 1, 0, stream>>>(ws, E);
  k_scatter<<<(B + 255) / 256, 256, 0, stream>>>(ids, ws, B, E);

  const int maxBlocks = (B + ROWS - 1) / ROWS + (E - 1);
  adapter_kernel<<<maxBlocks, NTHR, SMEM_BYTES, stream>>>(
      x, W1, b1, W2, b2, out, ws, B, E);
}
